// PAM_11252814316235
// MI455X (gfx1250) — matrix operand order repaired and checked by reading, not compiled, not run on hardware
//
#include <hip/hip_runtime.h>
#include <hip/hip_bf16.h>

typedef __attribute__((ext_vector_type(16))) _Float16 v16h;
typedef __attribute__((ext_vector_type(8)))  float    v8f;
typedef _Float16 half_t;

#define B_  4
#define C_  256
#define NF_ 32
#define N_  4096   // 64*64

// ---------------------------------------------------------------------------
// Q/K projection: q[b,n,f] = sum_c Wq[f,c]*x[b,c,n] + bq[f]  -> f16 row-major
// ---------------------------------------------------------------------------
__global__ void pam_proj_qk(const float* __restrict__ x,
                            const float* __restrict__ Wq, const float* __restrict__ bq,
                            const float* __restrict__ Wk, const float* __restrict__ bk,
                            half_t* __restrict__ Qh, half_t* __restrict__ Kh) {
    const int b = blockIdx.x >> 4;                       // 16 blocks of 256 n / batch
    const int n = ((blockIdx.x & 15) << 8) + threadIdx.x;
    float qa[NF_], ka[NF_];
#pragma unroll
    for (int f = 0; f < NF_; ++f) { qa[f] = 0.f; ka[f] = 0.f; }
    const float* xb = x + (size_t)b * C_ * N_ + n;
    for (int c = 0; c < C_; ++c) {
        const float xv = xb[(size_t)c * N_];             // coalesced across lanes
#pragma unroll
        for (int f = 0; f < NF_; ++f) {                  // Wq/Wk are wave-uniform (s_load)
            qa[f] = fmaf(Wq[f * C_ + c], xv, qa[f]);
            ka[f] = fmaf(Wk[f * C_ + c], xv, ka[f]);
        }
    }
    half_t* qrow = Qh + ((size_t)b * N_ + n) * NF_;
    half_t* krow = Kh + ((size_t)b * N_ + n) * NF_;
#pragma unroll
    for (int f = 0; f < NF_; ++f) {
        qrow[f] = (half_t)(qa[f] + bq[f]);
        krow[f] = (half_t)(ka[f] + bk[f]);
    }
}

// ---------------------------------------------------------------------------
// V projection: v[b,d,n] = sum_c Wv[d,c]*x[b,c,n] + bv[d]  -> f16, d-major
// ---------------------------------------------------------------------------
__global__ void pam_proj_v(const float* __restrict__ x,
                           const float* __restrict__ Wv, const float* __restrict__ bv,
                           half_t* __restrict__ Vh) {
    int idx = blockIdx.x;
    const int t = idx & 3;  idx >>= 2;                   // N_/1024 == 4
    const int d = idx % C_; idx /= C_;
    const int b = idx;
    const int n = t * 1024 + threadIdx.x * 4;
    const float* xb = x + (size_t)b * C_ * N_ + n;
    const float* wv = Wv + (size_t)d * C_;               // wave-uniform
    float4 acc = {0.f, 0.f, 0.f, 0.f};
    for (int c = 0; c < C_; ++c) {
        const float w = wv[c];
        const float4 xv = *(const float4*)(xb + (size_t)c * N_);
        acc.x = fmaf(w, xv.x, acc.x);
        acc.y = fmaf(w, xv.y, acc.y);
        acc.z = fmaf(w, xv.z, acc.z);
        acc.w = fmaf(w, xv.w, acc.w);
    }
    const float bb = bv[d];
    half_t* o = Vh + ((size_t)b * C_ + d) * N_ + n;
    o[0] = (half_t)(acc.x + bb);
    o[1] = (half_t)(acc.y + bb);
    o[2] = (half_t)(acc.z + bb);
    o[3] = (half_t)(acc.w + bb);
}

// ---------------------------------------------------------------------------
// Fused flash attention + residual. One wave per 32-row i-tile (2 WMMA tiles).
// Per 32-wide j-block:
//   S(32x32) = Q_i(32x32) Kj^T         (4x v_wmma_f32_16x16x32_f16)
//   full-wave online softmax: each of the 32 lanes owns one of the 32 rows
//   O(32x256) += P(32x32) Vj(32x256)   (32x v_wmma, V B-tile loaded once)
// ---------------------------------------------------------------------------
__global__ void __launch_bounds__(32) pam_flash(
        const half_t* __restrict__ Qh, const half_t* __restrict__ Kh,
        const half_t* __restrict__ Vh, const float* __restrict__ x,
        const float* __restrict__ gamma, float* __restrict__ out) {
    __shared__ __align__(16) float  s_lds[32 * 32];      // S tile (f32), rows 0-31
    __shared__ __align__(16) half_t p_lds[32 * 32];      // P tile (f16, row-major)
    __shared__ __align__(16) float  stat_lds[32];        // per-row alpha / 1/l

    const int lane   = threadIdx.x;
    const int row16  = lane & 15;
    const int half16 = lane >> 4;
    const int b  = blockIdx.x >> 7;                      // 128 i-tiles / batch
    const int i0 = (blockIdx.x & 127) << 5;              // 32 rows per wave

    // Q tiles in A-layout: lane = row (i0|i0+16)+row16; K-chunks split by lane half
    const half_t* qbase0 = Qh + ((size_t)b * N_ + i0 + row16) * NF_;
    const half_t* qbase1 = qbase0 + 16 * NF_;
    v16h qa0, qa1;
    *(float4*)&qa0       = *(const float4*)(qbase0 + 8 * half16);
    *((float4*)&qa0 + 1) = *(const float4*)(qbase0 + 16 + 8 * half16);
    *(float4*)&qa1       = *(const float4*)(qbase1 + 8 * half16);
    *((float4*)&qa1 + 1) = *(const float4*)(qbase1 + 16 + 8 * half16);

    v8f acc0[16], acc1[16];
    const v8f zeroc = {};
#pragma unroll
    for (int dc = 0; dc < 16; ++dc) { acc0[dc] = zeroc; acc1[dc] = zeroc; }

    float m_i = -INFINITY;   // this lane's row (row index == lane, 0..31)
    float l_i = 0.f;

    const half_t* kbase = Kh + (size_t)b * N_ * NF_;
    const half_t* vbase = Vh + (size_t)b * C_ * N_;

    for (int j0 = 0; j0 < N_; j0 += 32) {
        // K^T B-tiles: lane col j = j0(+16)+row16, f = 8*half16..+7 then 16+8*half16..+7
        v16h kb0, kb1;
        const half_t* kr0 = kbase + (size_t)(j0 + row16) * NF_ + 8 * half16;
        *(float4*)&kb0       = *(const float4*)(kr0);
        *((float4*)&kb0 + 1) = *(const float4*)(kr0 + 16);
        const half_t* kr1 = kr0 + 16 * NF_;
        *(float4*)&kb1       = *(const float4*)(kr1);
        *((float4*)&kb1 + 1) = *(const float4*)(kr1 + 16);

        v8f s00 = __builtin_amdgcn_wmma_f32_16x16x32_f16(false, qa0, false, kb0,
                                                         (short)0, zeroc, false, false);
        v8f s01 = __builtin_amdgcn_wmma_f32_16x16x32_f16(false, qa0, false, kb1,
                                                         (short)0, zeroc, false, false);
        v8f s10 = __builtin_amdgcn_wmma_f32_16x16x32_f16(false, qa1, false, kb0,
                                                         (short)0, zeroc, false, false);
        v8f s11 = __builtin_amdgcn_wmma_f32_16x16x32_f16(false, qa1, false, kb1,
                                                         (short)0, zeroc, false, false);
        // stage S (C-layout -> row-major LDS): VGPR r holds M=r+8*half16, N=row16
#pragma unroll
        for (int r = 0; r < 8; ++r) {
            const int m = r + 8 * half16;
            s_lds[m * 32 + row16]             = s00[r];
            s_lds[m * 32 + 16 + row16]        = s01[r];
            s_lds[(16 + m) * 32 + row16]      = s10[r];
            s_lds[(16 + m) * 32 + 16 + row16] = s11[r];
        }
        __syncthreads();

        {   // full-wave softmax row pass: lane owns row `lane`
            const float* srow = s_lds + lane * 32;
            float tmax = m_i;
#pragma unroll
            for (int e = 0; e < 32; ++e) tmax = fmaxf(tmax, srow[e]);
            const float alpha = __expf(m_i - tmax);
            float psum = 0.f;
            half_t* prow = p_lds + lane * 32;
#pragma unroll
            for (int e = 0; e < 32; ++e) {
                const float p = __expf(srow[e] - tmax);
                psum += p;
                prow[e] = (half_t)p;
            }
            l_i = l_i * alpha + psum;
            m_i = tmax;
            stat_lds[lane] = alpha;
        }
        __syncthreads();

        // P tiles back as A-layout f16
        v16h pa0, pa1;
        const half_t* prow0 = p_lds + row16 * 32;
        const half_t* prow1 = p_lds + (16 + row16) * 32;
        *(float4*)&pa0       = *(const float4*)(prow0 + 8 * half16);
        *((float4*)&pa0 + 1) = *(const float4*)(prow0 + 16 + 8 * half16);
        *(float4*)&pa1       = *(const float4*)(prow1 + 8 * half16);
        *((float4*)&pa1 + 1) = *(const float4*)(prow1 + 16 + 8 * half16);

        float af0[8], af1[8];
#pragma unroll
        for (int r = 0; r < 8; ++r) {
            af0[r] = stat_lds[r + 8 * half16];
            af1[r] = stat_lds[16 + r + 8 * half16];
        }

#pragma unroll
        for (int dc = 0; dc < 16; ++dc) {
            // V B-tile loaded ONCE, feeds both i-tiles' WMMAs
            const half_t* vr = vbase + (size_t)(dc * 16 + row16) * N_ + j0 + 8 * half16;
            v16h vb;
            *(float4*)&vb       = *(const float4*)(vr);
            *((float4*)&vb + 1) = *(const float4*)(vr + 16);
            v8f a0 = acc0[dc], a1 = acc1[dc];
#pragma unroll
            for (int r = 0; r < 8; ++r) { a0[r] *= af0[r]; a1[r] *= af1[r]; }
            acc0[dc] = __builtin_amdgcn_wmma_f32_16x16x32_f16(false, pa0, false, vb,
                                                              (short)0, a0, false, false);
            acc1[dc] = __builtin_amdgcn_wmma_f32_16x16x32_f16(false, pa1, false, vb,
                                                              (short)0, a1, false, false);
        }
        __syncthreads();                                 // protect LDS reuse
    }

    // epilogue: out = gamma * (O / l) + x
    stat_lds[lane] = 1.0f / l_i;
    __syncthreads();
    float lf0[8], lf1[8];
#pragma unroll
    for (int r = 0; r < 8; ++r) {
        lf0[r] = stat_lds[r + 8 * half16];
        lf1[r] = stat_lds[16 + r + 8 * half16];
    }
    const float g = gamma[0];

#pragma unroll
    for (int dc = 0; dc < 16; ++dc) {
        const int d = dc * 16 + row16;
        const size_t base0 = ((size_t)b * C_ + d) * N_ + i0 + 8 * half16;   // 16B aligned
        const size_t base1 = base0 + 16;
        const float4 x00 = *(const float4*)(x + base0);
        const float4 x01 = *(const float4*)(x + base0 + 4);
        const float4 x10 = *(const float4*)(x + base1);
        const float4 x11 = *(const float4*)(x + base1 + 4);
        float4 o00, o01, o10, o11;
        o00.x = g * (acc0[dc][0] * lf0[0]) + x00.x;
        o00.y = g * (acc0[dc][1] * lf0[1]) + x00.y;
        o00.z = g * (acc0[dc][2] * lf0[2]) + x00.z;
        o00.w = g * (acc0[dc][3] * lf0[3]) + x00.w;
        o01.x = g * (acc0[dc][4] * lf0[4]) + x01.x;
        o01.y = g * (acc0[dc][5] * lf0[5]) + x01.y;
        o01.z = g * (acc0[dc][6] * lf0[6]) + x01.z;
        o01.w = g * (acc0[dc][7] * lf0[7]) + x01.w;
        o10.x = g * (acc1[dc][0] * lf1[0]) + x10.x;
        o10.y = g * (acc1[dc][1] * lf1[1]) + x10.y;
        o10.z = g * (acc1[dc][2] * lf1[2]) + x10.z;
        o10.w = g * (acc1[dc][3] * lf1[3]) + x10.w;
        o11.x = g * (acc1[dc][4] * lf1[4]) + x11.x;
        o11.y = g * (acc1[dc][5] * lf1[5]) + x11.y;
        o11.z = g * (acc1[dc][6] * lf1[6]) + x11.z;
        o11.w = g * (acc1[dc][7] * lf1[7]) + x11.w;
        *(float4*)(out + base0)     = o00;
        *(float4*)(out + base0 + 4) = o01;
        *(float4*)(out + base1)     = o10;
        *(float4*)(out + base1 + 4) = o11;
    }
}

// ---------------------------------------------------------------------------
extern "C" void kernel_launch(void* const* d_in, const int* in_sizes, int n_in,
                              void* d_out, int out_size, void* d_ws, size_t ws_size,
                              hipStream_t stream) {
    const float* x     = (const float*)d_in[0];
    const float* Wq    = (const float*)d_in[1];
    const float* bq    = (const float*)d_in[2];
    const float* Wk    = (const float*)d_in[3];
    const float* bk    = (const float*)d_in[4];
    const float* Wv    = (const float*)d_in[5];
    const float* bv    = (const float*)d_in[6];
    const float* gamma = (const float*)d_in[7];
    float* out = (float*)d_out;

    half_t* Qh = (half_t*)d_ws;                          // 1 MB
    half_t* Kh = Qh + (size_t)B_ * N_ * NF_;             // 1 MB
    half_t* Vh = Kh + (size_t)B_ * N_ * NF_;             // 8 MB (d-major)

    pam_proj_qk<<<B_ * (N_ / 256), 256, 0, stream>>>(x, Wq, bq, Wk, bk, Qh, Kh);
    pam_proj_v <<<B_ * C_ * (N_ / 1024), 256, 0, stream>>>(x, Wv, bv, Vh);
    pam_flash  <<<B_ * (N_ / 32), 32, 0, stream>>>(Qh, Kh, Vh, x, gamma, out);
}
